// RenderMAEEncoder_28956669509887
// MI455X (gfx1250) — matrix operand order repaired and checked by reading, not compiled, not run on hardware
//
#include <hip/hip_runtime.h>
#include <hip/hip_bf16.h>

#define BB 64
#define NN 2048
#define GG 64
#define KK 32
#define DD 384
#define HH 6
#define HDD 64
#define MLPH 1536
#define DEPTH_ 12
#define EPSF 1e-5f

typedef _Float16 half_t;
typedef __attribute__((ext_vector_type(16))) _Float16 v16h;
typedef __attribute__((ext_vector_type(8)))  _Float16 v8h;
typedef __attribute__((ext_vector_type(2)))  _Float16 v2h;
typedef __attribute__((ext_vector_type(8)))  float    v8f;

__device__ __forceinline__ float gelu_exact(float x) {
    return 0.5f * x * (1.0f + erff(x * 0.70710678118654752f));
}

// ---------------------------------------------------------------------------
// WMMA fragment loaders (CDNA5 ISA 7.12.2, wave32).
// All loaders read contiguous 16B chunks per lane -> ds_load_b128.
// ---------------------------------------------------------------------------
__device__ __forceinline__ v16h load_frag_a_row(const half_t* base, int lda, int lane) {
    const int m  = lane & 15;
    const int hs = (lane >> 4) & 1;
    const half_t* p = base + m * lda;
    const v8h lo = *(const v8h*)(p + hs * 16);
    const v8h hi = *(const v8h*)(p + hs * 16 + 8);
    return __builtin_shufflevector(lo, hi, 0, 1, 2, 3, 4, 5, 6, 7,
                                           8, 9, 10, 11, 12, 13, 14, 15);
}

__device__ __forceinline__ v16h load_frag_b_sw(const half_t* tileBase, int lane) {
    const v8h lo = *(const v8h*)(tileBase + lane * 16);
    const v8h hi = *(const v8h*)(tileBase + lane * 16 + 8);
    return __builtin_shufflevector(lo, hi, 0, 1, 2, 3, 4, 5, 6, 7,
                                           8, 9, 10, 11, 12, 13, 14, 15);
}

__device__ __forceinline__ v16h load_frag_bt_row(const half_t* base, int ldb, int lane) {
    const int n  = lane & 15;
    const int kb = (lane >> 4) ? 16 : 0;
    const half_t* p = base + n * ldb + kb;
    const v8h lo = *(const v8h*)(p);
    const v8h hi = *(const v8h*)(p + 8);
    return __builtin_shufflevector(lo, hi, 0, 1, 2, 3, 4, 5, 6, 7,
                                           8, 9, 10, 11, 12, 13, 14, 15);
}

// ---------------------------------------------------------------------------
// Farthest point sampling: one block per batch
// ---------------------------------------------------------------------------
__global__ __launch_bounds__(256) void fps_kernel(const float* __restrict__ pts,
                                                  float* __restrict__ centers) {
    const int b = blockIdx.x, t = threadIdx.x;
    __shared__ float dist[NN];
    __shared__ float rval[256];
    __shared__ int   ridx[256];
    __shared__ int   lastIdx;
    const float* P = pts + (size_t)b * NN * 3;
    for (int n = t; n < NN; n += 256) dist[n] = 1e10f;
    if (t == 0) lastIdx = 0;
    __syncthreads();
    for (int it = 0; it < GG; ++it) {
        const int li = lastIdx;
        const float px = P[li * 3 + 0], py = P[li * 3 + 1], pz = P[li * 3 + 2];
        if (t == 0) {
            float* c = centers + ((size_t)b * GG + it) * 3;
            c[0] = px; c[1] = py; c[2] = pz;
        }
        if (it == GG - 1) break;
        float best = -1e30f; int bi = 0;
        for (int n = t; n < NN; n += 256) {
            const float dx = P[n * 3] - px, dy = P[n * 3 + 1] - py, dz = P[n * 3 + 2] - pz;
            const float dn = fminf(dist[n], dx * dx + dy * dy + dz * dz);
            dist[n] = dn;
            if (dn > best) { best = dn; bi = n; }
        }
        rval[t] = best; ridx[t] = bi;
        __syncthreads();
        for (int s = 128; s > 0; s >>= 1) {
            if (t < s) {
                if (rval[t + s] > rval[t] ||
                    (rval[t + s] == rval[t] && ridx[t + s] < ridx[t])) {
                    rval[t] = rval[t + s]; ridx[t] = ridx[t + s];
                }
            }
            __syncthreads();
        }
        if (t == 0) lastIdx = ridx[0];
        __syncthreads();
    }
}

// ---------------------------------------------------------------------------
// kNN (K smallest d2) + gather + subtract center: one block per (b,g)
// ---------------------------------------------------------------------------
__global__ __launch_bounds__(256) void knn_kernel(const float* __restrict__ pts,
                                                  const float* __restrict__ centers,
                                                  float* __restrict__ grouped) {
    const int bg = blockIdx.x, b = bg / GG, t = threadIdx.x;
    __shared__ float d2s[NN];
    __shared__ float rval[256];
    __shared__ int   ridx[256];
    const float* P = pts + (size_t)b * NN * 3;
    const float* c = centers + (size_t)bg * 3;
    const float cx = c[0], cy = c[1], cz = c[2];
    for (int n = t; n < NN; n += 256) {
        const float dx = P[n * 3] - cx, dy = P[n * 3 + 1] - cy, dz = P[n * 3 + 2] - cz;
        d2s[n] = dx * dx + dy * dy + dz * dz;
    }
    __syncthreads();
    for (int kk = 0; kk < KK; ++kk) {
        float best = 3e38f; int bi = NN;
        for (int n = t; n < NN; n += 256) {
            const float v = d2s[n];
            if (v < best) { best = v; bi = n; }
        }
        rval[t] = best; ridx[t] = bi;
        __syncthreads();
        for (int s = 128; s > 0; s >>= 1) {
            if (t < s) {
                if (rval[t + s] < rval[t] ||
                    (rval[t + s] == rval[t] && ridx[t + s] < ridx[t])) {
                    rval[t] = rval[t + s]; ridx[t] = ridx[t + s];
                }
            }
            __syncthreads();
        }
        if (t == 0) {
            const int n = ridx[0];
            float* go = grouped + ((size_t)bg * KK + kk) * 3;
            go[0] = P[n * 3] - cx; go[1] = P[n * 3 + 1] - cy; go[2] = P[n * 3 + 2] - cz;
            d2s[n] = 3e38f;
        }
        __syncthreads();
    }
}

// ---------------------------------------------------------------------------
// Mini-PointNet encoder (3->64->128->384, relu, relu, maxpool over K)
// ---------------------------------------------------------------------------
__global__ __launch_bounds__(128) void encoder_kernel(
    const float* __restrict__ grouped,
    const float* __restrict__ w1, const float* __restrict__ b1,
    const float* __restrict__ w2, const float* __restrict__ b2,
    const float* __restrict__ w3, const float* __restrict__ b3,
    float* __restrict__ tok) {
    const int bg = blockIdx.x, t = threadIdx.x;
    __shared__ float pt[3];
    __shared__ float h1[64];
    __shared__ float h2[128];
    float mx[3] = {-3e38f, -3e38f, -3e38f};
    for (int k = 0; k < KK; ++k) {
        if (t < 3) pt[t] = grouped[((size_t)bg * KK + k) * 3 + t];
        __syncthreads();
        if (t < 64) {
            float a = b1[t];
            for (int j = 0; j < 3; ++j) a += pt[j] * w1[j * 64 + t];
            h1[t] = fmaxf(a, 0.0f);
        }
        __syncthreads();
        {
            float a = b2[t];
            for (int j = 0; j < 64; ++j) a += h1[j] * w2[j * 128 + t];
            h2[t] = fmaxf(a, 0.0f);
        }
        __syncthreads();
#pragma unroll
        for (int o = 0; o < 3; ++o) {
            const int col = t + o * 128;
            float a = b3[col];
            for (int j = 0; j < 128; ++j) a += h2[j] * w3[j * DD + col];
            mx[o] = fmaxf(mx[o], a);
        }
        __syncthreads();
    }
#pragma unroll
    for (int o = 0; o < 3; ++o) tok[(size_t)bg * DD + t + o * 128] = mx[o];
}

// ---------------------------------------------------------------------------
// Positional MLP: gelu(c @ W1 + b1) @ W2 + b2, one block per (b,g)
// ---------------------------------------------------------------------------
__global__ __launch_bounds__(128) void pos_kernel(
    const float* __restrict__ centers,
    const float* __restrict__ w1, const float* __restrict__ b1,
    const float* __restrict__ w2, const float* __restrict__ b2,
    float* __restrict__ pos) {
    const int bg = blockIdx.x, t = threadIdx.x;
    __shared__ float c3[3];
    __shared__ float hh[128];
    if (t < 3) c3[t] = centers[(size_t)bg * 3 + t];
    __syncthreads();
    {
        float a = b1[t];
        for (int j = 0; j < 3; ++j) a += c3[j] * w1[j * 128 + t];
        hh[t] = gelu_exact(a);
    }
    __syncthreads();
#pragma unroll
    for (int o = 0; o < 3; ++o) {
        const int col = t + o * 128;
        float a = b2[col];
        for (int j = 0; j < 128; ++j) a += hh[j] * w2[j * DD + col];
        pos[(size_t)bg * DD + col] = a;
    }
}

// ---------------------------------------------------------------------------
// (optional add pos) + LayerNorm; out16 (f16 activations) or out32 (final)
// ---------------------------------------------------------------------------
__global__ __launch_bounds__(128) void ln_kernel(
    float* __restrict__ x, const float* __restrict__ pos,
    const float* __restrict__ gam, const float* __restrict__ bet,
    half_t* __restrict__ out16, float* __restrict__ out32) {
    const int row = blockIdx.x, t = threadIdx.x;
    __shared__ float red[128];
    float* xr = x + (size_t)row * DD;
    float vals[3];
#pragma unroll
    for (int o = 0; o < 3; ++o) {
        const int i = t + o * 128;
        float v = xr[i];
        if (pos) { v += pos[(size_t)row * DD + i]; xr[i] = v; }
        vals[o] = v;
    }
    red[t] = vals[0] + vals[1] + vals[2];
    __syncthreads();
    for (int s = 64; s > 0; s >>= 1) { if (t < s) red[t] += red[t + s]; __syncthreads(); }
    const float mean = red[0] * (1.0f / DD);
    __syncthreads();
    const float d0 = vals[0] - mean, d1 = vals[1] - mean, d2 = vals[2] - mean;
    red[t] = d0 * d0 + d1 * d1 + d2 * d2;
    __syncthreads();
    for (int s = 64; s > 0; s >>= 1) { if (t < s) red[t] += red[t + s]; __syncthreads(); }
    const float rstd = rsqrtf(red[0] * (1.0f / DD) + EPSF);
#pragma unroll
    for (int o = 0; o < 3; ++o) {
        const int i = t + o * 128;
        const float y = (vals[o] - mean) * rstd * gam[i] + bet[i];
        if (out16) out16[(size_t)row * DD + i] = (half_t)y;
        else       out32[(size_t)row * DD + i] = y;
    }
}

// ---------------------------------------------------------------------------
// WMMA GEMM: out(M x N) = epilogue(A(M x K) @ W(K x N) + bias)
// 256 threads (8 waves), block tile 64 x 128, wave -> 32 x 32 slab
// (2 A-frags x 2 B-frags = 4 WMMAs per 32-deep K step, operands reused).
// Double-buffered LDS (ping-pong): prefetch kt+1 into registers during the
// WMMAs of kt, store to the alternate buffer, one barrier per K step.
// B staged pre-swizzled to fragment order with pair-packed b32 stores.
// ---------------------------------------------------------------------------
#define GEMM_BIAS      0
#define GEMM_BIAS_GELU 1
#define GEMM_RESID     2

__global__ __launch_bounds__(256) void gemm_wmma(
    const half_t* __restrict__ A, const half_t* __restrict__ W,
    const float* __restrict__ bias,
    half_t* __restrict__ out16, float* __restrict__ out32,
    int Kdim, int Ndim, int mode) {
    __shared__ __align__(32) half_t As[2][64][40];       // row-major + pad
    __shared__ __align__(32) half_t Bsw[2][8][32][16];   // [16-col tile][lane][j]
    const int tid  = threadIdx.x;
    const int lane = tid & 31;
    const int wave = tid >> 5;
    const int wr   = wave >> 2;   // 0..1 : 32-row half
    const int wc   = wave & 3;    // 0..3 : 32-col slab
    const int row0 = blockIdx.x * 64;
    const int col0 = blockIdx.y * 128;

    // staging coordinates
    const int ar = tid >> 2;                 // A row 0..63
    const int ac = (tid & 3) * 8;            // A col chunk
    const int br = (tid >> 4) * 2;           // even B k-row 0..30
    const int bc0 = (tid & 15) * 8;          // B col 0..120 (within one tile)
    const int btile  = bc0 >> 4;
    const int blane0 = ((br >> 4) << 4) | (bc0 & 15);
    const int bj     = br & 15;              // even fragment index

    v8h pa, pb0, pb1;
    auto glob_load = [&](int kt) {
        pa  = *(const v8h*)(A + (size_t)(row0 + ar) * Kdim + kt * 32 + ac);
        pb0 = *(const v8h*)(W + (size_t)(kt * 32 + br) * Ndim + col0 + bc0);
        pb1 = *(const v8h*)(W + (size_t)(kt * 32 + br + 1) * Ndim + col0 + bc0);
    };
    auto lds_store = [&](int buf) {
        *(v8h*)(&As[buf][ar][ac]) = pa;
#pragma unroll
        for (int e = 0; e < 8; ++e) {
            v2h p = {pb0[e], pb1[e]};
            *(v2h*)(&Bsw[buf][btile][blane0 + e][bj]) = p;  // paired b32 store
        }
    };

    v8f acc00 = {}, acc01 = {}, acc10 = {}, acc11 = {};
    const int nk = Kdim >> 5;
    glob_load(0);
    lds_store(0);
    __syncthreads();
    for (int kt = 0; kt < nk; ++kt) {
        const int cur = kt & 1;
        if (kt + 1 < nk) glob_load(kt + 1);
        const v16h a0 = load_frag_a_row(&As[cur][wr * 32][0], 40, lane);
        const v16h a1 = load_frag_a_row(&As[cur][wr * 32 + 16][0], 40, lane);
        const v16h b0 = load_frag_b_sw(&Bsw[cur][wc * 2][0][0], lane);
        const v16h b1 = load_frag_b_sw(&Bsw[cur][wc * 2 + 1][0][0], lane);
        acc00 = __builtin_amdgcn_wmma_f32_16x16x32_f16(false, a0, false, b0,
                                                       (short)0, acc00, false, false);
        acc01 = __builtin_amdgcn_wmma_f32_16x16x32_f16(false, a0, false, b1,
                                                       (short)0, acc01, false, false);
        acc10 = __builtin_amdgcn_wmma_f32_16x16x32_f16(false, a1, false, b0,
                                                       (short)0, acc10, false, false);
        acc11 = __builtin_amdgcn_wmma_f32_16x16x32_f16(false, a1, false, b1,
                                                       (short)0, acc11, false, false);
        if (kt + 1 < nk) lds_store(cur ^ 1);
        __syncthreads();
    }
    const int n  = lane & 15;
    const int m0 = (lane >> 4) * 8;
#pragma unroll
    for (int sub = 0; sub < 2; ++sub) {
#pragma unroll
        for (int ts = 0; ts < 2; ++ts) {
            const v8f acc = sub ? (ts ? acc11 : acc10) : (ts ? acc01 : acc00);
#pragma unroll
            for (int r = 0; r < 8; ++r) {
                const int grow = row0 + wr * 32 + sub * 16 + m0 + r;
                const int gcol = col0 + wc * 32 + ts * 16 + n;
                const float v = acc[r] + bias[gcol];
                const size_t o = (size_t)grow * Ndim + gcol;
                if (mode == GEMM_BIAS)           out16[o] = (half_t)v;
                else if (mode == GEMM_BIAS_GELU) out16[o] = (half_t)gelu_exact(v);
                else                             out32[o] += v;
            }
        }
    }
}

// ---------------------------------------------------------------------------
// Attention, one block per (b,h): S=QK^T*scale, softmax, O=P@V (all WMMA).
// Q/K staged with 16B vector copies; V staged pre-swizzled (pair-packed).
// ---------------------------------------------------------------------------
__global__ __launch_bounds__(128) void attn_wmma(const half_t* __restrict__ qkv,
                                                 half_t* __restrict__ o16) {
    __shared__ __align__(32) half_t Qs[64][72];
    __shared__ __align__(32) half_t Ks[64][72];
    __shared__ __align__(32) half_t Vsw[2][4][32][16];  // [kc][nt][lane][j]
    __shared__ float  Sf[64][66];
    __shared__ __align__(32) half_t Ps[64][72];

    const int bh = blockIdx.x;
    const int b = bh / HH, h = bh % HH;
    const int tid = threadIdx.x;
    const int lane = tid & 31;
    const int w = tid >> 5;  // wave -> 16-row block

    // Q/K: 512 16B chunks each (4 per thread)
    for (int i = tid; i < 512; i += 128) {
        const int g = i >> 3, c0 = (i & 7) * 8;
        const half_t* rowp = qkv + (size_t)(b * GG + g) * (3 * DD) + h * HDD;
        *(v8h*)(&Qs[g][c0]) = *(const v8h*)(rowp + c0);
        *(v8h*)(&Ks[g][c0]) = *(const v8h*)(rowp + DD + c0);
    }
    // V: 256 pair-chunks (2 per thread), scattered into fragment order
    for (int i = tid; i < 256; i += 128) {
        const int g0 = (i >> 3) * 2, c0 = (i & 7) * 8;
        const v8h v0 = *(const v8h*)(qkv + (size_t)(b * GG + g0) * (3 * DD) +
                                     2 * DD + h * HDD + c0);
        const v8h v1 = *(const v8h*)(qkv + (size_t)(b * GG + g0 + 1) * (3 * DD) +
                                     2 * DD + h * HDD + c0);
        const int kc = g0 >> 5, tile = c0 >> 4;
        const int lane0 = (((g0 >> 4) & 1) << 4) | (c0 & 15);
        const int j = g0 & 15;  // even
#pragma unroll
        for (int e = 0; e < 8; ++e) {
            v2h p = {v0[e], v1[e]};
            *(v2h*)(&Vsw[kc][tile][lane0 + e][j]) = p;
        }
    }
    __syncthreads();

    const float scale = 0.125f;  // HD^-0.5
    const int n  = lane & 15;
    const int m0 = (lane >> 4) * 8;
#pragma unroll
    for (int nt = 0; nt < 4; ++nt) {
        v8f acc = {};
#pragma unroll
        for (int kc = 0; kc < 2; ++kc) {
            const v16h af = load_frag_a_row(&Qs[w * 16][kc * 32], 72, lane);
            const v16h bf = load_frag_bt_row(&Ks[nt * 16][kc * 32], 72, lane);
            acc = __builtin_amdgcn_wmma_f32_16x16x32_f16(false, af, false, bf,
                                                         (short)0, acc, false, false);
        }
#pragma unroll
        for (int r = 0; r < 8; ++r) Sf[w * 16 + m0 + r][nt * 16 + n] = acc[r] * scale;
    }
    __syncthreads();
    if (tid < 64) {
        float mx = -3e38f;
        for (int j = 0; j < 64; ++j) mx = fmaxf(mx, Sf[tid][j]);
        float s = 0.0f;
        for (int j = 0; j < 64; ++j) {
            const float e = expf(Sf[tid][j] - mx);
            s += e;
            Ps[tid][j] = (half_t)e;
        }
        const float inv = 1.0f / s;
        for (int j = 0; j < 64; ++j) Ps[tid][j] = (half_t)((float)Ps[tid][j] * inv);
    }
    __syncthreads();
#pragma unroll
    for (int nt = 0; nt < 4; ++nt) {
        v8f acc = {};
#pragma unroll
        for (int kc = 0; kc < 2; ++kc) {
            const v16h af = load_frag_a_row(&Ps[w * 16][kc * 32], 72, lane);
            const v16h bf = load_frag_b_sw(&Vsw[kc][nt][0][0], lane);
            acc = __builtin_amdgcn_wmma_f32_16x16x32_f16(false, af, false, bf,
                                                         (short)0, acc, false, false);
        }
#pragma unroll
        for (int r = 0; r < 8; ++r) {
            const int g = w * 16 + m0 + r;
            o16[(size_t)(b * GG + g) * DD + h * HDD + nt * 16 + n] = (half_t)acc[r];
        }
    }
}

// ---------------------------------------------------------------------------
__global__ __launch_bounds__(256) void f32_to_f16_kernel(const float* __restrict__ in,
                                                         half_t* __restrict__ out,
                                                         size_t nElem) {
    size_t i = (size_t)blockIdx.x * blockDim.x + threadIdx.x;
    const size_t stride = (size_t)gridDim.x * blockDim.x;
    for (; i < nElem; i += stride) out[i] = (half_t)in[i];
}

// ---------------------------------------------------------------------------
extern "C" void kernel_launch(void* const* d_in, const int* in_sizes, int n_in,
                              void* d_out, int out_size, void* d_ws, size_t ws_size,
                              hipStream_t stream) {
    (void)in_sizes; (void)n_in; (void)out_size; (void)ws_size;
    const float* vis_pts = (const float*)d_in[0];
    const float* enc_w1  = (const float*)d_in[1];
    const float* enc_b1  = (const float*)d_in[2];
    const float* enc_w2  = (const float*)d_in[3];
    const float* enc_b2  = (const float*)d_in[4];
    const float* enc_w3  = (const float*)d_in[5];
    const float* enc_b3  = (const float*)d_in[6];
    const float* pos_w1  = (const float*)d_in[7];
    const float* pos_b1  = (const float*)d_in[8];
    const float* pos_w2  = (const float*)d_in[9];
    const float* pos_b2  = (const float*)d_in[10];
    const float* ln1_s   = (const float*)d_in[11];
    const float* ln1_b   = (const float*)d_in[12];
    const float* qkv_w   = (const float*)d_in[13];
    const float* qkv_b   = (const float*)d_in[14];
    const float* proj_w  = (const float*)d_in[15];
    const float* proj_b  = (const float*)d_in[16];
    const float* ln2_s   = (const float*)d_in[17];
    const float* ln2_b   = (const float*)d_in[18];
    const float* fc1_w   = (const float*)d_in[19];
    const float* fc1_b   = (const float*)d_in[20];
    const float* fc2_w   = (const float*)d_in[21];
    const float* fc2_b   = (const float*)d_in[22];
    const float* lnf_s   = (const float*)d_in[23];
    const float* lnf_b   = (const float*)d_in[24];

    float* x_out   = (float*)d_out;                       // (B,G,D)
    float* centers = x_out + (size_t)BB * GG * DD;        // (B,G,3)

    char* ws = (char*)d_ws;
    size_t off = 0;
    auto alloc = [&](size_t bytes) -> void* {
        void* p = ws + off;
        off += (bytes + 255) & ~(size_t)255;
        return p;
    };
    float*  grouped = (float*)alloc((size_t)BB * GG * KK * 3 * 4);
    float*  x       = (float*)alloc((size_t)BB * GG * DD * 4);
    float*  pos     = (float*)alloc((size_t)BB * GG * DD * 4);
    half_t* hh16    = (half_t*)alloc((size_t)BB * GG * DD * 2);
    half_t* qkv16   = (half_t*)alloc((size_t)BB * GG * 3 * DD * 2);
    half_t* o16     = (half_t*)alloc((size_t)BB * GG * DD * 2);
    half_t* hid16   = (half_t*)alloc((size_t)BB * GG * MLPH * 2);
    half_t* wqkv16  = (half_t*)alloc((size_t)DEPTH_ * DD * 3 * DD * 2);
    half_t* wproj16 = (half_t*)alloc((size_t)DEPTH_ * DD * DD * 2);
    half_t* wfc116  = (half_t*)alloc((size_t)DEPTH_ * DD * MLPH * 2);
    half_t* wfc216  = (half_t*)alloc((size_t)DEPTH_ * MLPH * DD * 2);

    // per-call weight conversions (fp16 working set stays L2-resident)
    f32_to_f16_kernel<<<1024, 256, 0, stream>>>(qkv_w, wqkv16, (size_t)DEPTH_ * DD * 3 * DD);
    f32_to_f16_kernel<<<1024, 256, 0, stream>>>(proj_w, wproj16, (size_t)DEPTH_ * DD * DD);
    f32_to_f16_kernel<<<1024, 256, 0, stream>>>(fc1_w, wfc116, (size_t)DEPTH_ * DD * MLPH);
    f32_to_f16_kernel<<<1024, 256, 0, stream>>>(fc2_w, wfc216, (size_t)DEPTH_ * MLPH * DD);

    fps_kernel<<<BB, 256, 0, stream>>>(vis_pts, centers);
    knn_kernel<<<BB * GG, 256, 0, stream>>>(vis_pts, centers, grouped);
    encoder_kernel<<<BB * GG, 128, 0, stream>>>(grouped, enc_w1, enc_b1, enc_w2, enc_b2,
                                                enc_w3, enc_b3, x);
    pos_kernel<<<BB * GG, 128, 0, stream>>>(centers, pos_w1, pos_b1, pos_w2, pos_b2, pos);

    const int Mrows = BB * GG;  // 4096
    for (int l = 0; l < DEPTH_; ++l) {
        ln_kernel<<<Mrows, 128, 0, stream>>>(x, pos, ln1_s + (size_t)l * DD,
                                             ln1_b + (size_t)l * DD, hh16, nullptr);
        {
            dim3 grid(Mrows / 64, (3 * DD) / 128);
            gemm_wmma<<<grid, 256, 0, stream>>>(hh16, wqkv16 + (size_t)l * DD * 3 * DD,
                                                qkv_b + (size_t)l * 3 * DD, qkv16, nullptr,
                                                DD, 3 * DD, GEMM_BIAS);
        }
        attn_wmma<<<BB * HH, 128, 0, stream>>>(qkv16, o16);
        {
            dim3 grid(Mrows / 64, DD / 128);
            gemm_wmma<<<grid, 256, 0, stream>>>(o16, wproj16 + (size_t)l * DD * DD,
                                                proj_b + (size_t)l * DD, nullptr, x,
                                                DD, DD, GEMM_RESID);
        }
        ln_kernel<<<Mrows, 128, 0, stream>>>(x, nullptr, ln2_s + (size_t)l * DD,
                                             ln2_b + (size_t)l * DD, hh16, nullptr);
        {
            dim3 grid(Mrows / 64, MLPH / 128);
            gemm_wmma<<<grid, 256, 0, stream>>>(hh16, wfc116 + (size_t)l * DD * MLPH,
                                                fc1_b + (size_t)l * MLPH, hid16, nullptr,
                                                DD, MLPH, GEMM_BIAS_GELU);
        }
        {
            dim3 grid(Mrows / 64, DD / 128);
            gemm_wmma<<<grid, 256, 0, stream>>>(hid16, wfc216 + (size_t)l * MLPH * DD,
                                                fc2_b + (size_t)l * DD, nullptr, x,
                                                MLPH, DD, GEMM_RESID);
        }
    }
    ln_kernel<<<Mrows, 128, 0, stream>>>(x, nullptr, lnf_s, lnf_b, nullptr, x_out);
}
